// GMEmbedder2ConvSortPoolReadout_15178414424423
// MI455X (gfx1250) — matrix operand order repaired and checked by reading, compiled, not run on hardware
//
#include <hip/hip_runtime.h>

typedef __attribute__((ext_vector_type(16))) _Float16 v16h;
typedef __attribute__((ext_vector_type(8)))  _Float16 v8h;
typedef __attribute__((ext_vector_type(8)))  float    v8f;

#define B_GRAPHS   8
#define N_NODES    12500
#define TOTAL_N    (B_GRAPHS * N_NODES)       // 100000
#define E_TOTAL    1600000
#define K_POOL     64
#define C1         128
#define C2         32
#define OUT_PER_G  (K_POOL * (C1 + C2))       // 10240
#define R2_OFF     (K_POOL * C1)              // 8192
#define EPS_GN     1e-5f

__device__ __forceinline__ float leaky(float x) { return x >= 0.f ? x : 0.01f * x; }

// ---------------------------------------------------------------- utilities
__global__ void zero_f32(float* __restrict__ p, long n) {
    long i = (long)blockIdx.x * blockDim.x + threadIdx.x;
    if (i < n) p[i] = 0.f;
}

__global__ void f32_to_f16(const float* __restrict__ x, _Float16* __restrict__ y, int n) {
    int i = blockIdx.x * blockDim.x + threadIdx.x;
    if (i < n) y[i] = (_Float16)x[i];
}

__global__ void degrees_kernel(const int* __restrict__ src, const int* __restrict__ dst,
                               float* __restrict__ degO, float* __restrict__ degI, int E) {
    int i = blockIdx.x * blockDim.x + threadIdx.x;
    if (i < E) {
        atomicAdd(&degO[src[i]], 1.f);
        atomicAdd(&degI[dst[i]], 1.f);
    }
}

// ------------------------------------------------- WMMA tiled dense GEMM
// Y[rows, NOUT] = (X[rows, K] * rsqrt(max(degO,1))[:,None]) @ Wh[K, NOUT]
// 256 threads = 8 waves; each wave owns a 16-row M tile; loops over N tiles.
// A staged row-major (16B-aligned rows) -> fragment = 2x ds_load_b128.
// W staged PRE-SWIZZLED into per-lane fragment order -> fragment = 2x ds_load_b128.
template <int K, int NOUT>
__global__ void gemm_wmma(const float* __restrict__ X, const _Float16* __restrict__ Wh,
                          const float* __restrict__ degO, float* __restrict__ Y, int rows) {
    constexpr int MT   = 128;
    constexpr int APAD = 8;                       // keeps rows 16B-aligned, breaks bank conflicts
    constexpr int NT   = NOUT / 16;
    constexpr int KC   = K / 32;
    __shared__ alignas(16) _Float16 As[MT * (K + APAD)];
    __shared__ alignas(16) _Float16 Ws[NT * KC * 32 * 16];   // == K*NOUT halves

    const int tid     = threadIdx.x;
    const int rowBase = blockIdx.x * MT;

    // stage A tile (scaled, f16) into LDS
    for (int idx = tid; idx < MT * K; idx += 256) {
        int r = idx / K, c = idx % K;
        int grow = rowBase + r;
        float v = 0.f;
        if (grow < rows) {
            float d = degO[grow]; d = d < 1.f ? 1.f : d;
            v = X[(size_t)grow * K + c] * rsqrtf(d);
        }
        As[r * (K + APAD) + c] = (_Float16)v;
    }
    // stage B pre-swizzled: Ws[(((nt*KC+kc)*32 + lane)<<4) + j] holds the j-th half of
// lane's v16h fragment for tile (nt,kc):  B[kc*32 + (lane>>4)*8 + (j&7) + (j>>3)*16][nt*16 + (lane&15)]
    for (int idx = tid; idx < NT * KC * 32 * 16; idx += 256) {
        int j    = idx & 15;
        int lane = (idx >> 4) & 31;
        int kc   = (idx >> 9) % KC;
        int nt   = (idx >> 9) / KC;
        int kk   = kc * 32 + (lane >> 4) * 8 + (j & 7) + (j >> 3) * 16;
        int nn   = nt * 16 + (lane & 15);
        Ws[idx] = Wh[kk * NOUT + nn];
    }
    __syncthreads();

    const int wave = tid >> 5, lane = tid & 31;
    const int lm = lane & 15;        // M (for A) / N (for B,C) sub-index
    const int hi = lane >> 4;        // half-wave selector
    const int mRow = wave * 16;

    for (int nt = 0; nt < NT; ++nt) {
        v8f acc = {};
        #pragma unroll
        for (int kc = 0; kc < KC; ++kc) {
            // A fragment, 16-bit A 16x32 layout (ISA 7.12.2):
            // lanes 0-15: K = kc*32 + {0..7, 16..23}; lanes 16-31: +8
            const _Float16* arow = &As[(mRow + lm) * (K + APAD) + kc * 32 + hi * 8];
            v8h alo = *(const v8h*)arow;          // ds_load_b128
            v8h ahi = *(const v8h*)(arow + 16);   // ds_load_b128
            v16h a = __builtin_shufflevector(alo, ahi,
                         0,1,2,3,4,5,6,7,8,9,10,11,12,13,14,15);
            // B fragment: 32 contiguous bytes per lane (2x ds_load_b128)
            v16h b = *(const v16h*)&Ws[(((nt * KC + kc) * 32) + lane) << 4];
            acc = __builtin_amdgcn_wmma_f32_16x16x32_f16(
                false, a, false, b, (short)0, acc, false, false);
        }
        // C/D layout: VGPR r -> M = r + 8*hi, N = lm
        const int orow = rowBase + mRow + hi * 8;
        float* yp = &Y[(size_t)orow * NOUT + nt * 16 + lm];
        if (orow + 7 < rows) {                     // uniform fast path: no exec churn
            #pragma unroll
            for (int r = 0; r < 8; ++r) yp[(size_t)r * NOUT] = acc[r];
        } else {
            #pragma unroll
            for (int r = 0; r < 8; ++r)
                if (orow + r < rows) yp[(size_t)r * NOUT] = acc[r];
        }
    }
}

// ------------------------------------------------- edge scatter (SpMM)
// wave per edge; C==128: each lane moves 16 contiguous bytes (global_load_b128)
template <int C>
__global__ void spmm_scatter(const float* __restrict__ H, const int* __restrict__ src,
                             const int* __restrict__ dst, const float* __restrict__ ew,
                             float* __restrict__ out, int E) {
    int wave = (blockIdx.x * blockDim.x + threadIdx.x) >> 5;
    int lane = threadIdx.x & 31;
    if (wave >= E) return;
    int s = src[wave], d = dst[wave];
    float w = ew[wave];
    if (wave + 4 < E)   // cover gather latency out of L2 (global_prefetch_b8)
        __builtin_prefetch(&H[(size_t)src[wave + 4] * C], 0, 1);
    if constexpr (C == 128) {
        const float4 h = *(const float4*)&H[(size_t)s * C + lane * 4];
        float* op = &out[(size_t)d * C + lane * 4];
        atomicAdd(op + 0, h.x * w);
        atomicAdd(op + 1, h.y * w);
        atomicAdd(op + 2, h.z * w);
        atomicAdd(op + 3, h.w * w);
    } else {
        atomicAdd(&out[(size_t)d * C + lane], H[(size_t)s * C + lane] * w);
    }
}

__global__ void scale_rows(float* __restrict__ H, const float* __restrict__ degI,
                           long nTot, int C) {
    long i = (long)blockIdx.x * blockDim.x + threadIdx.x;
    if (i < nTot) {
        float d = degI[i / C]; d = d < 1.f ? 1.f : d;
        H[i] *= rsqrtf(d);
    }
}

// ------------------------------------------------- GraphNorm (one-pass stats)
__global__ void gnorm_stats(const float* __restrict__ H, float* __restrict__ stats,
                            const float* __restrict__ alpha, int C) {
    int g = blockIdx.x / C, c = blockIdx.x % C;
    __shared__ float s1[256], s2[256];
    float sx = 0.f, sxx = 0.f;
    const float* base = H + (size_t)g * N_NODES * C + c;
    for (int i = threadIdx.x; i < N_NODES; i += 256) {
        float x = base[(size_t)i * C];
        sx += x; sxx += x * x;
    }
    s1[threadIdx.x] = sx; s2[threadIdx.x] = sxx;
    __syncthreads();
    for (int s = 128; s > 0; s >>= 1) {
        if (threadIdx.x < s) { s1[threadIdx.x] += s1[threadIdx.x + s];
                               s2[threadIdx.x] += s2[threadIdx.x + s]; }
        __syncthreads();
    }
    if (threadIdx.x == 0) {
        float m   = s1[0] / (float)N_NODES;
        float ex2 = s2[0] / (float)N_NODES;
        float a   = alpha[c];
        // var((x - a*m)) = E[x^2] - m^2 * (2a - a^2)
        float var = ex2 - m * m * (2.f * a - a * a);
        stats[2 * (g * C + c)]     = m;
        stats[2 * (g * C + c) + 1] = rsqrtf(var + EPS_GN);
    }
}

__global__ void gnorm_apply(float* __restrict__ H, const float* __restrict__ stats,
                            const float* __restrict__ gamma, const float* __restrict__ beta,
                            const float* __restrict__ alpha, long nTot, int C) {
    long i = (long)blockIdx.x * blockDim.x + threadIdx.x;
    if (i >= nTot) return;
    long node = i / C;
    int  c    = (int)(i % C);
    int  g    = (int)(node / N_NODES);
    float m    = stats[2 * (g * C + c)];
    float istd = stats[2 * (g * C + c) + 1];
    float y = gamma[c] * (H[i] - alpha[c] * m) * istd + beta[c];
    H[i] = leaky(y);
}

// ------------------------------------------------- sort-pool pieces
__global__ void row_max_key(const float* __restrict__ H, float* __restrict__ key,
                            int n, int C) {
    int wave = (blockIdx.x * blockDim.x + threadIdx.x) >> 5;
    int lane = threadIdx.x & 31;
    if (wave >= n) return;
    float m = -3.4e38f;
    for (int c = lane; c < C; c += 32) m = fmaxf(m, H[(size_t)wave * C + c]);
    for (int off = 16; off >= 1; off >>= 1) m = fmaxf(m, __shfl_xor(m, off, 32));
    if (lane == 0) key[wave] = m;
}

__global__ void topk_kernel(const float* __restrict__ key, int* __restrict__ idxOut) {
    int g = blockIdx.x;
    __shared__ float bv[256];
    __shared__ int   bi[256];
    __shared__ unsigned char msk[N_NODES];
    for (int i = threadIdx.x; i < N_NODES; i += 256) msk[i] = 0;
    __syncthreads();
    const float* kg = key + (size_t)g * N_NODES;
    for (int k = 0; k < K_POOL; ++k) {
        float best = -3.4e38f; int besti = N_NODES;
        for (int i = threadIdx.x; i < N_NODES; i += 256) {
            if (!msk[i]) {
                float v = kg[i];
                if (v > best || (v == best && i < besti)) { best = v; besti = i; }
            }
        }
        bv[threadIdx.x] = best; bi[threadIdx.x] = besti;
        __syncthreads();
        for (int s = 128; s > 0; s >>= 1) {
            if (threadIdx.x < s) {
                float v2 = bv[threadIdx.x + s]; int i2 = bi[threadIdx.x + s];
                if (v2 > bv[threadIdx.x] ||
                    (v2 == bv[threadIdx.x] && i2 < bi[threadIdx.x])) {
                    bv[threadIdx.x] = v2; bi[threadIdx.x] = i2;
                }
            }
            __syncthreads();
        }
        if (threadIdx.x == 0) { idxOut[g * K_POOL + k] = bi[0]; msk[bi[0]] = 1; }
        __syncthreads();
    }
}

// one block per selected node; blockDim == C; bitonic ascending sort + final leaky
__global__ void sort_write(const float* __restrict__ H, const int* __restrict__ idx,
                           float* __restrict__ out, int C, int outOff) {
    int b = blockIdx.x / K_POOL, k = blockIdx.x % K_POOL;
    long node = (long)b * N_NODES + idx[b * K_POOL + k];
    __shared__ float sd[128];
    int t = threadIdx.x;
    sd[t] = H[node * C + t];
    __syncthreads();
    for (int size = 2; size <= C; size <<= 1) {
        for (int stride = size >> 1; stride > 0; stride >>= 1) {
            int p = t ^ stride;
            float a = sd[t], bb = sd[p];
            bool up = ((t & size) == 0);
            float mn = fminf(a, bb), mx = fmaxf(a, bb);
            float res = up ? (t < p ? mn : mx) : (t < p ? mx : mn);
            __syncthreads();
            sd[t] = res;
            __syncthreads();
        }
    }
    out[(size_t)b * OUT_PER_G + outOff + k * C + t] = leaky(sd[t]);
}

// ---------------------------------------------------------------- launcher
extern "C" void kernel_launch(void* const* d_in, const int* in_sizes, int n_in,
                              void* d_out, int out_size, void* d_ws, size_t ws_size,
                              hipStream_t stream) {
    const float* features = (const float*)d_in[0];
    const float* ew       = (const float*)d_in[1];
    const float* W1       = (const float*)d_in[2];
    const float* W2       = (const float*)d_in[3];
    const float* gamma1   = (const float*)d_in[4];
    const float* beta1    = (const float*)d_in[5];
    const float* alpha1   = (const float*)d_in[6];
    const float* gamma2   = (const float*)d_in[7];
    const float* beta2    = (const float*)d_in[8];
    const float* alpha2   = (const float*)d_in[9];
    const int*   src      = (const int*)d_in[10];
    const int*   dst      = (const int*)d_in[11];
    float*       out      = (float*)d_out;

    float* ws   = (float*)d_ws;
    float* bufA = ws;                              // 12,800,000 f32 (h projected)
    float* bufB = bufA + (size_t)TOTAL_N * C1;     // 12,800,000 f32 (h1)
    float* bufC = bufB + (size_t)TOTAL_N * C1;     //  3,200,000 f32 (h2)
    float* degO = bufC + (size_t)TOTAL_N * C2;     //    100,000
    float* degI = degO + TOTAL_N;                  //    100,000
    float* key  = degI + TOTAL_N;                  //    100,000
    float* stats= key + TOTAL_N;                   //      2,048
    int*   tkid = (int*)(stats + 2048);            //        512 ints
    _Float16* W1h = (_Float16*)(tkid + 512);       //      8,192 halves
    _Float16* W2h = W1h + 64 * 128;                //      4,096 halves

    const int T = 256;
    auto blks = [](long n, int t) { return (int)((n + t - 1) / t); };

    // degrees + weight conversion
    zero_f32<<<blks(2 * TOTAL_N, T), T, 0, stream>>>(degO, 2 * (long)TOTAL_N);
    degrees_kernel<<<blks(E_TOTAL, T), T, 0, stream>>>(src, dst, degO, degI, E_TOTAL);
    f32_to_f16<<<blks(64 * 128, T), T, 0, stream>>>(W1, W1h, 64 * 128);
    f32_to_f16<<<blks(128 * 32, T), T, 0, stream>>>(W2, W2h, 128 * 32);

    // ---- layer 1 ----
    gemm_wmma<64, 128><<<blks(TOTAL_N, 128), T, 0, stream>>>(features, W1h, degO, bufA, TOTAL_N);
    zero_f32<<<blks((long)TOTAL_N * C1, T), T, 0, stream>>>(bufB, (long)TOTAL_N * C1);
    spmm_scatter<C1><<<blks((long)E_TOTAL * 32, T), T, 0, stream>>>(bufA, src, dst, ew, bufB, E_TOTAL);
    scale_rows<<<blks((long)TOTAL_N * C1, T), T, 0, stream>>>(bufB, degI, (long)TOTAL_N * C1, C1);
    gnorm_stats<<<B_GRAPHS * C1, T, 0, stream>>>(bufB, stats, alpha1, C1);
    gnorm_apply<<<blks((long)TOTAL_N * C1, T), T, 0, stream>>>(bufB, stats, gamma1, beta1, alpha1,
                                                               (long)TOTAL_N * C1, C1);
    row_max_key<<<blks((long)TOTAL_N * 32, T), T, 0, stream>>>(bufB, key, TOTAL_N, C1);
    topk_kernel<<<B_GRAPHS, T, 0, stream>>>(key, tkid);
    sort_write<<<B_GRAPHS * K_POOL, C1, 0, stream>>>(bufB, tkid, out, C1, 0);

    // ---- layer 2 ----
    gemm_wmma<128, 32><<<blks(TOTAL_N, 128), T, 0, stream>>>(bufB, W2h, degO, bufA, TOTAL_N);
    zero_f32<<<blks((long)TOTAL_N * C2, T), T, 0, stream>>>(bufC, (long)TOTAL_N * C2);
    spmm_scatter<C2><<<blks((long)E_TOTAL * 32, T), T, 0, stream>>>(bufA, src, dst, ew, bufC, E_TOTAL);
    scale_rows<<<blks((long)TOTAL_N * C2, T), T, 0, stream>>>(bufC, degI, (long)TOTAL_N * C2, C2);
    gnorm_stats<<<B_GRAPHS * C2, T, 0, stream>>>(bufC, stats, alpha2, C2);
    gnorm_apply<<<blks((long)TOTAL_N * C2, T), T, 0, stream>>>(bufC, stats, gamma2, beta2, alpha2,
                                                               (long)TOTAL_N * C2, C2);
    row_max_key<<<blks((long)TOTAL_N * 32, T), T, 0, stream>>>(bufC, key, TOTAL_N, C2);
    topk_kernel<<<B_GRAPHS, T, 0, stream>>>(key, tkid);
    sort_write<<<B_GRAPHS * K_POOL, C2, 0, stream>>>(bufC, tkid, out, C2, R2_OFF);
}
